// ScaledDotProductAttention_MultiHead_24781961298462
// MI455X (gfx1250) — matrix operand order repaired and checked by reading, not compiled, not run on hardware
//
#include <hip/hip_runtime.h>

// ---------------------------------------------------------------------------
// Fused multi-head attention for B=4, H=16, S=1024, D=64 (fp32 in/out).
//   scores = (Q K^T)/sqrt(D) -> mask -> softmax -> attn (written to d_out)
//   out    = attn @ V                            (written to d_out)
// One wave32 per 16-row q-tile. Score strip in LDS (f32); probabilities
// re-packed in place as f16 pairs for the PV pass; V tiles staged in LDS
// cooperatively (packed f16 k-pairs) once per block. WMMA f16 -> f32.
// ---------------------------------------------------------------------------

typedef __attribute__((ext_vector_type(16))) _Float16 v16h;
typedef __attribute__((ext_vector_type(8)))  float    v8f;

constexpr int B_ = 4;
constexpr int H_ = 16;
constexpr int S_ = 1024;
constexpr int D_ = 64;

constexpr int QROWS = 16;        // q rows per wave
constexpr int WAVES = 4;         // waves per workgroup
constexpr int SROW  = S_ + 16;   // padded LDS row stride in floats (65 banks)
constexpr int VBUF_U32 = 32 * D_ / 2;   // 32x64 V tile as packed half2 = 1024 u32
constexpr size_t LDS_BYTES =
    (size_t)WAVES * QROWS * SROW * sizeof(float) + VBUF_U32 * sizeof(unsigned);

__device__ __forceinline__ float wave_max32(float v) {
#pragma unroll
  for (int o = 16; o > 0; o >>= 1) v = fmaxf(v, __shfl_xor(v, o, 32));
  return v;
}
__device__ __forceinline__ float wave_sum32(float v) {
#pragma unroll
  for (int o = 16; o > 0; o >>= 1) v += __shfl_xor(v, o, 32);
  return v;
}

// A-fragment K offset map for V_WMMA_*_16X16X32_F16 (16-bit A, 16x32):
// VGPR j holds K = kb + {0,2,4,6,16,18,20,22}[j] and +1  (kb = 0 or 8).
__device__ __forceinline__ int amap(int j) { return (j < 4) ? 2 * j : 16 + 2 * (j - 4); }

__device__ __forceinline__ unsigned pack2h(float a, float b) {
  union { unsigned u; _Float16 h[2]; } c;
  c.h[0] = (_Float16)a; c.h[1] = (_Float16)b;
  return c.u;
}

__global__ void __launch_bounds__(WAVES * 32, 1)
sdpa_fused_kernel(const float* __restrict__ Q,
                  const float* __restrict__ K,
                  const float* __restrict__ V,
                  const int*   __restrict__ mask,   // [B,1,1,S]
                  float* __restrict__ out,          // [B,H,S,D]
                  float* __restrict__ attn)         // [B,H,S,S]
{
  extern __shared__ float smem[];

  const int tid  = threadIdx.x;
  const int lane = tid & 31;
  const int wave = tid >> 5;
  const int b    = blockIdx.z;
  const int h    = blockIdx.y;
  const int bh   = b * H_ + h;
  const int q0   = blockIdx.x * (WAVES * QROWS) + wave * QROWS;

  float*    lds  = smem + (size_t)wave * QROWS * SROW;  // this wave's strip
  unsigned* vbuf = (unsigned*)(smem + (size_t)WAVES * QROWS * SROW);

  const int m   = lane & 15;              // A-row / B-col / C-col index
  const int kbA = (lane < 16) ? 0 : 8;    // A fragment K base
  const int kbB = (lane < 16) ? 0 : 16;   // B fragment K base
  const int kbH = kbB >> 1;               // B fragment K-pair base (0 or 8)
  const int mb  = (lane < 16) ? 0 : 8;    // C/D fragment row base

  // ---- Q fragments for this q-tile: d = [0,32) and [32,64) --------------
  const float* qrow = Q + ((size_t)bh * S_ + q0 + m) * D_;
  v16h aq0, aq1;
#pragma unroll
  for (int j = 0; j < 8; ++j) {
    const int k  = kbB + 2 * j;
    const float2 f0 = *(const float2*)(qrow + k);
    const float2 f1 = *(const float2*)(qrow + 32 + k);
    aq0[2 * j] = (_Float16)f0.x;  aq0[2 * j + 1] = (_Float16)f0.y;
    aq1[2 * j] = (_Float16)f1.x;  aq1[2 * j + 1] = (_Float16)f1.y;
  }

  // ---- per-lane mask bits: bit (4c+t) covers column c*128 + lane*4 + t --
  unsigned mbits = 0u;
#pragma unroll
  for (int c = 0; c < 8; ++c) {
    const int4 mv = *(const int4*)(mask + b * S_ + c * 128 + lane * 4);
    mbits |= (mv.x != 0 ? 1u : 0u) << (4 * c + 0);
    mbits |= (mv.y != 0 ? 1u : 0u) << (4 * c + 1);
    mbits |= (mv.z != 0 ? 1u : 0u) << (4 * c + 2);
    mbits |= (mv.w != 0 ? 1u : 0u) << (4 * c + 3);
  }

  // ---- phase 1: score strip (Q K^T)/8 into LDS --------------------------
  for (int kt = 0; kt < S_ / 16; ++kt) {
    const float* krow = K + ((size_t)bh * S_ + kt * 16 + m) * D_;  // key row = col n
    v16h bk0, bk1;
#pragma unroll
    for (int j = 0; j < 8; ++j) {
      const int k  = kbB + 2 * j;
      const float2 f0 = *(const float2*)(krow + k);
      const float2 f1 = *(const float2*)(krow + 32 + k);
      bk0[2 * j] = (_Float16)f0.x;  bk0[2 * j + 1] = (_Float16)f0.y;
      bk1[2 * j] = (_Float16)f1.x;  bk1[2 * j + 1] = (_Float16)f1.y;
    }
    v8f c = {0.f, 0.f, 0.f, 0.f, 0.f, 0.f, 0.f, 0.f};
    c = __builtin_amdgcn_wmma_f32_16x16x32_f16(false, aq0, false, bk0,
                                               (short)0, c, false, false);
    c = __builtin_amdgcn_wmma_f32_16x16x32_f16(false, aq1, false, bk1,
                                               (short)0, c, false, false);
#pragma unroll
    for (int j = 0; j < 8; ++j)
      lds[(mb + j) * SROW + kt * 16 + m] = c[j] * 0.125f;   // 1/sqrt(64)
  }

  // ---- phase 2: masked softmax per row; write attn; pack f16 into LDS ---
  for (int r = 0; r < QROWS; ++r) {
    float* rowp = lds + r * SROW;
    float s[32];
#pragma unroll
    for (int c = 0; c < 8; ++c) {       // vector loads: 8 x ds_load_b128
      const float4 f = *(const float4*)(rowp + c * 128 + lane * 4);
      s[4 * c + 0] = ((mbits >> (4 * c + 0)) & 1u) ? f.x : -100000.0f;
      s[4 * c + 1] = ((mbits >> (4 * c + 1)) & 1u) ? f.y : -100000.0f;
      s[4 * c + 2] = ((mbits >> (4 * c + 2)) & 1u) ? f.z : -100000.0f;
      s[4 * c + 3] = ((mbits >> (4 * c + 3)) & 1u) ? f.w : -100000.0f;
    }
    float mx = s[0];
#pragma unroll
    for (int i = 1; i < 32; ++i) mx = fmaxf(mx, s[i]);
    mx = wave_max32(mx);

    float acc = 0.f;
#pragma unroll
    for (int i = 0; i < 32; ++i) { s[i] = __expf(s[i] - mx); acc += s[i]; }
    const float inv = 1.0f / wave_sum32(acc);

    float*    arow = attn + ((size_t)bh * S_ + q0 + r) * S_;
    unsigned* prow = (unsigned*)rowp;   // packed f16 probs, uint index = col/2
#pragma unroll
    for (int c = 0; c < 8; ++c) {
      float4 p;
      p.x = ((mbits >> (4 * c + 0)) & 1u) ? s[4 * c + 0] * inv : 0.0f;
      p.y = ((mbits >> (4 * c + 1)) & 1u) ? s[4 * c + 1] * inv : 0.0f;
      p.z = ((mbits >> (4 * c + 2)) & 1u) ? s[4 * c + 2] * inv : 0.0f;
      p.w = ((mbits >> (4 * c + 3)) & 1u) ? s[4 * c + 3] * inv : 0.0f;
      *(float4*)(arow + c * 128 + lane * 4) = p;            // global_store_b128
      uint2 pk;
      pk.x = pack2h(p.x, p.y);
      pk.y = pack2h(p.z, p.w);
      *(uint2*)(prow + c * 64 + lane * 2) = pk;             // ds_store_b64
    }
  }

  // ---- phase 3: out = attn @ V ------------------------------------------
  // V tile (32 rows x 64 cols) staged per block in LDS as packed f16
  // k-pairs: vbuf[r*64 + n] = half2(V[2r][n], V[2r+1][n]).
  v8f acc[4];
#pragma unroll
  for (int nt = 0; nt < 4; ++nt)
    acc[nt] = (v8f){0.f, 0.f, 0.f, 0.f, 0.f, 0.f, 0.f, 0.f};

  const unsigned* myrowU = (const unsigned*)(lds + m * SROW);
  const int vr  = tid >> 3;         // 0..15 : k-pair row this thread stages
  const int vn0 = (tid & 7) * 8;    // 0..56 : 8-column chunk this thread stages

  for (int kc = 0; kc < S_ / 32; ++kc) {
    // cooperative coalesced V loads (two adjacent rows, 8 cols each)
    const float* vrow = V + ((size_t)bh * S_ + kc * 32 + 2 * vr) * D_ + vn0;
    const float4 a0 = *(const float4*)(vrow);
    const float4 a1 = *(const float4*)(vrow + 4);
    const float4 b0 = *(const float4*)(vrow + D_);
    const float4 b1 = *(const float4*)(vrow + D_ + 4);
    uint4 o0, o1;
    o0.x = pack2h(a0.x, b0.x);  o0.y = pack2h(a0.y, b0.y);
    o0.z = pack2h(a0.z, b0.z);  o0.w = pack2h(a0.w, b0.w);
    o1.x = pack2h(a1.x, b1.x);  o1.y = pack2h(a1.y, b1.y);
    o1.z = pack2h(a1.z, b1.z);  o1.w = pack2h(a1.w, b1.w);

    __syncthreads();                           // previous tile fully consumed
    *(uint4*)(vbuf + vr * 64 + vn0)     = o0;  // ds_store_b128
    *(uint4*)(vbuf + vr * 64 + vn0 + 4) = o1;
    __syncthreads();                           // tile ready

    union { unsigned u[8]; v16h v; } ap;
#pragma unroll
    for (int j = 0; j < 8; ++j) {
      const int k = kbB + 2 * j;               // even; pair (k, k+1)
      ap.u[j] = myrowU[kc * 16 + (k >> 1)];    // ds_load_b32, pre-packed
    }
#pragma unroll
    for (int nt = 0; nt < 4; ++nt) {
      union { unsigned u[8]; v16h v; } bp;
#pragma unroll
      for (int j = 0; j < 8; ++j)
        bp.u[j] = vbuf[(kbH + j) * 64 + nt * 16 + m];   // ds_load_b32
      acc[nt] = __builtin_amdgcn_wmma_f32_16x16x32_f16(false, ap.v, false, bp.v,
                                                       (short)0, acc[nt],
                                                       false, false);
    }
  }

#pragma unroll
  for (int nt = 0; nt < 4; ++nt)
#pragma unroll
    for (int j = 0; j < 8; ++j)
      out[((size_t)bh * S_ + q0 + mb + j) * D_ + nt * 16 + m] = acc[nt][j];
}

extern "C" void kernel_launch(void* const* d_in, const int* in_sizes, int n_in,
                              void* d_out, int out_size, void* d_ws, size_t ws_size,
                              hipStream_t stream) {
  (void)in_sizes; (void)n_in; (void)out_size; (void)d_ws; (void)ws_size;

  const float* Q    = (const float*)d_in[0];
  const float* K    = (const float*)d_in[1];
  const float* V    = (const float*)d_in[2];
  const int*   mask = (const int*)d_in[3];

  float* out  = (float*)d_out;                                   // [B,H,S,D]
  float* attn = (float*)d_out + (size_t)B_ * H_ * S_ * D_;       // [B,H,S,S]

  dim3 grid(S_ / (WAVES * QROWS), H_, B_);   // (16, 16, 4)
  dim3 block(WAVES * 32, 1, 1);              // 4 wave32s

  sdpa_fused_kernel<<<grid, block, LDS_BYTES, stream>>>(Q, K, V, mask, out, attn);
}
